// slg_v1_5282809774460
// MI455X (gfx1250) — compile-verified, matrix operand order checked by reading, not run on hardware
//
#include <hip/hip_runtime.h>
#include <hip/hip_bf16.h>

typedef __attribute__((ext_vector_type(16))) _Float16 v16h;
typedef __attribute__((ext_vector_type(8)))  float    v8f;

#define DEV __device__ __forceinline__

// ---- problem constants (fixed by the reference) ----
static constexpr int NNODES = 50000;
static constexpr int EEDGES = 800000;
static constexpr int EUNIQ  = 400000;
static constexpr int L2NODE = 131072;
static constexpr int ELEDGE = 1048576;
static constexpr int NGRAPH = 512;
static constexpr int HID    = 128;     // H
static constexpr int NHEAD  = 4;
static constexpr int SEQ    = 6;
static constexpr int NLAYER = 2;
static constexpr int NTOK   = L2NODE * SEQ;   // 786432

// =====================================================================
//  WMMA GEMM:  C[M,N] = A[M,K] @ W[N,K]^T + bias[N]   (MODE 1 -> relu)
//  f32 in memory, converted to f16 in registers, f32 accumulate.
//  Each wave computes a 16x32 tile (two 16x16 WMMA accumulators) so the
//  A-fragment is loaded once and used twice. K is a compile-time constant
//  -> fully unrolled k-loop, no divergent guards, offsets folded.
// =====================================================================
template <bool FULL>
DEV v16h ld_frag(const float* __restrict__ row, int k0, int hlf) {
  v16h r;
  const int c0 = k0 + hlf * 8;
  {
    float4 a = *(const float4*)(row + c0);
    float4 b = *(const float4*)(row + c0 + 4);
    r[0]=(_Float16)a.x; r[1]=(_Float16)a.y; r[2]=(_Float16)a.z; r[3]=(_Float16)a.w;
    r[4]=(_Float16)b.x; r[5]=(_Float16)b.y; r[6]=(_Float16)b.z; r[7]=(_Float16)b.w;
  }
  if constexpr (FULL) {
    const int c1 = c0 + 16;
    float4 a = *(const float4*)(row + c1);
    float4 b = *(const float4*)(row + c1 + 4);
    r[8]=(_Float16)a.x;  r[9]=(_Float16)a.y;  r[10]=(_Float16)a.z; r[11]=(_Float16)a.w;
    r[12]=(_Float16)b.x; r[13]=(_Float16)b.y; r[14]=(_Float16)b.z; r[15]=(_Float16)b.w;
  } else {
    r[8]=(_Float16)0.f;  r[9]=(_Float16)0.f;  r[10]=(_Float16)0.f; r[11]=(_Float16)0.f;
    r[12]=(_Float16)0.f; r[13]=(_Float16)0.f; r[14]=(_Float16)0.f; r[15]=(_Float16)0.f;
  }
  return r;
}

template <int K, int MODE>
__global__ void __launch_bounds__(256) gemm_wmma(
    float* __restrict__ C, const float* __restrict__ A, const float* __restrict__ W,
    const float* __restrict__ bias, int M, int Ncols) {
  const int lane = threadIdx.x & 31;
  const int wid  = blockIdx.x * (blockDim.x >> 5) + (threadIdx.x >> 5);
  const int tilesN = Ncols >> 5;                 // 32-wide N tiles
  const int tiles  = (M >> 4) * tilesN;
  if (wid >= tiles) return;                      // whole wave exits together
  const int tm  = (wid / tilesN) << 4;
  const int tn  = (wid % tilesN) << 5;
  const int m   = lane & 15;
  const int hlf = lane >> 4;
  const float* arow  = A + (size_t)(tm + m) * K;
  const float* wrow0 = W + (size_t)(tn + m) * K;
  const float* wrow1 = W + (size_t)(tn + 16 + m) * K;
  v8f acc0, acc1;
  const float bv0 = bias ? bias[tn + m] : 0.f;
  const float bv1 = bias ? bias[tn + 16 + m] : 0.f;
#pragma unroll
  for (int r = 0; r < 8; ++r) { acc0[r] = bv0; acc1[r] = bv1; }
  constexpr bool FULL = (K % 32) == 0;
#pragma unroll
  for (int k0 = 0; k0 < K; k0 += 32) {
    v16h af = ld_frag<FULL>(arow,  k0, hlf);
    v16h w0 = ld_frag<FULL>(wrow0, k0, hlf);
    v16h w1 = ld_frag<FULL>(wrow1, k0, hlf);
    acc0 = __builtin_amdgcn_wmma_f32_16x16x32_f16(
        false, af, false, w0, (short)0, acc0, false, false);
    acc1 = __builtin_amdgcn_wmma_f32_16x16x32_f16(
        false, af, false, w1, (short)0, acc1, false, false);
  }
  const int n0 = tn + m;
  const int n1 = tn + 16 + m;
#pragma unroll
  for (int r = 0; r < 8; ++r) {
    const size_t rowoff = (size_t)(tm + hlf * 8 + r) * Ncols;
    float v0 = acc0[r], v1 = acc1[r];
    if (MODE == 1) { v0 = v0 > 0.f ? v0 : 0.f; v1 = v1 > 0.f ? v1 : 0.f; }
    C[rowoff + n0] = v0;
    C[rowoff + n1] = v1;
  }
}

// =====================================================================
//  Fills / elementwise
// =====================================================================
__global__ void fill_f32(float* p, size_t n, float v) {
  size_t i = (size_t)blockIdx.x * blockDim.x + threadIdx.x;
  if (i < n) p[i] = v;
}
__global__ void fill_u32(unsigned* p, size_t n, unsigned v) {
  size_t i = (size_t)blockIdx.x * blockDim.x + threadIdx.x;
  if (i < n) p[i] = v;
}
__global__ void relu_ip(float* p, size_t n) {
  size_t i = (size_t)blockIdx.x * blockDim.x + threadIdx.x;
  if (i < n) { float v = p[i]; p[i] = v > 0.f ? v : 0.f; }
}
__global__ void rsqrt_ip(float* p, int n) {
  int i = blockIdx.x * blockDim.x + threadIdx.x;
  if (i < n) { float v = p[i]; p[i] = v > 0.f ? rsqrtf(v) : 0.f; }
}

// =====================================================================
//  GCN pieces
// =====================================================================
__global__ void deg_scatter(float* __restrict__ deg, const int* __restrict__ dst, int nE) {
  int e = blockIdx.x * blockDim.x + threadIdx.x;
  if (e < nE) atomicAdd(&deg[dst[e]], 1.0f);
}
__global__ void gcn_init(float* __restrict__ Hd, const float* __restrict__ XW,
                         const float* __restrict__ dinv, const float* __restrict__ bias,
                         int nrows) {
  int i = blockIdx.x * blockDim.x + threadIdx.x;
  if (i >= nrows * HID) return;
  int r = i >> 7, c = i & 127;
  float dv = dinv[r];
  Hd[i] = dv * dv * XW[i] + bias[c];
}
__global__ void gcn_scatter(float* __restrict__ Hd, const float* __restrict__ XW,
                            const float* __restrict__ dinv, const int* __restrict__ src,
                            const int* __restrict__ dst, int nE) {
  int idx = blockIdx.x * blockDim.x + threadIdx.x;
  int e = idx >> 5;
  if (e >= nE) return;
  int c4 = (idx & 31) * 4;
  int s = src[e], d = dst[e];
  float nrm = dinv[s] * dinv[d];
  float4 xv = *(const float4*)(XW + (size_t)s * HID + c4);
  float* hp = Hd + (size_t)d * HID + c4;
  atomicAdd(hp + 0, nrm * xv.x);
  atomicAdd(hp + 1, nrm * xv.y);
  atomicAdd(hp + 2, nrm * xv.z);
  atomicAdd(hp + 3, nrm * xv.w);
}

// =====================================================================
//  Gather H0_set -> xs [L2,6,128]
// =====================================================================
__global__ void gather_set(float* __restrict__ xs, const float* __restrict__ H0,
                           const float* __restrict__ He, const int* __restrict__ m0,
                           const int* __restrict__ m1, const int* __restrict__ ue0,
                           const int* __restrict__ ue1) {
  int i = blockIdx.x * blockDim.x + threadIdx.x;
  if (i >= L2NODE * HID) return;
  int l = i >> 7, c = i & 127;
  int e = m0[l], f = m1[l];
  int u = ue0[e], v = ue1[e], s = ue0[f], t = ue1[f];
  float* o = xs + (size_t)l * (SEQ * HID);
  o[0 * HID + c] = H0[(size_t)u * HID + c];
  o[1 * HID + c] = H0[(size_t)v * HID + c];
  o[2 * HID + c] = H0[(size_t)s * HID + c];
  o[3 * HID + c] = H0[(size_t)t * HID + c];
  o[4 * HID + c] = He[(size_t)e * HID + c];
  o[5 * HID + c] = He[(size_t)f * HID + c];
}

// =====================================================================
//  Attention over 6-token sequences (per (seq, head) thread)
// =====================================================================
__global__ void attn_kernel(float* __restrict__ outp, const float* __restrict__ qkv) {
  int i = blockIdx.x * blockDim.x + threadIdx.x;
  if (i >= L2NODE * NHEAD) return;
  int l = i >> 2, h = i & 3;
  const float* base = qkv + (size_t)l * SEQ * 384;
  const float* q = base + h * 32;
  const float* k = base + 128 + h * 32;
  const float* v = base + 256 + h * 32;
  for (int s = 0; s < SEQ; ++s) {
    float sc[SEQ]; float mx = -3.4e38f;
    for (int t = 0; t < SEQ; ++t) {
      float d = 0.f;
      for (int dd = 0; dd < 32; ++dd) d += q[s * 384 + dd] * k[t * 384 + dd];
      sc[t] = d * 0.17677669529663687f;     // 1/sqrt(32)
      mx = sc[t] > mx ? sc[t] : mx;
    }
    float sum = 0.f;
    for (int t = 0; t < SEQ; ++t) { sc[t] = expf(sc[t] - mx); sum += sc[t]; }
    float inv = 1.f / sum;
    float* orow = outp + ((size_t)l * SEQ + s) * HID + h * 32;
    for (int dd = 0; dd < 32; ++dd) {
      float a = 0.f;
      for (int t = 0; t < SEQ; ++t) a += sc[t] * v[t * 384 + dd];
      orow[dd] = a * inv;
    }
  }
}

// =====================================================================
//  Row LayerNorm (optional residual, optional exact gelu), 128 cols,
//  one wave32 per row with shfl_xor reductions.
// =====================================================================
__global__ void __launch_bounds__(256) row_ln(
    float* __restrict__ x, const float* __restrict__ res,
    const float* __restrict__ g, const float* __restrict__ b,
    int rows, int gelu_flag) {
  const int lane = threadIdx.x & 31;
  const int row  = blockIdx.x * (blockDim.x >> 5) + (threadIdx.x >> 5);
  if (row >= rows) return;
  float* xr = x + (size_t)row * HID;
  float v[4]; float s = 0.f;
#pragma unroll
  for (int j = 0; j < 4; ++j) {
    int c = lane + 32 * j;
    float t = xr[c];
    if (res) t += res[(size_t)row * HID + c];
    v[j] = t; s += t;
  }
#pragma unroll
  for (int o = 16; o > 0; o >>= 1) s += __shfl_xor(s, o);
  const float mean = s * (1.f / 128.f);
  float var = 0.f;
#pragma unroll
  for (int j = 0; j < 4; ++j) { float d = v[j] - mean; var += d * d; }
#pragma unroll
  for (int o = 16; o > 0; o >>= 1) var += __shfl_xor(var, o);
  const float inv = rsqrtf(var * (1.f / 128.f) + 1e-5f);
#pragma unroll
  for (int j = 0; j < 4; ++j) {
    int c = lane + 32 * j;
    float y = (v[j] - mean) * inv * g[c] + b[c];
    if (gelu_flag) y = 0.5f * y * (1.f + erff(y * 0.70710678118654752f));
    xr[c] = y;
  }
}

// =====================================================================
//  Mean over 6 tokens, l2_batch, pooling
// =====================================================================
__global__ void mean6(float* __restrict__ H1o, const float* __restrict__ xs) {
  int i = blockIdx.x * blockDim.x + threadIdx.x;
  if (i >= L2NODE * HID) return;
  int l = i >> 7, c = i & 127;
  const float* p = xs + (size_t)l * (SEQ * HID);
  float s = p[c] + p[128 + c] + p[256 + c] + p[384 + c] + p[512 + c] + p[640 + c];
  H1o[i] = s * (1.f / 6.f);
}
__global__ void l2batch_kernel(int* __restrict__ lb, const int* __restrict__ batch,
                               const int* __restrict__ ue0, const int* __restrict__ m0) {
  int l = blockIdx.x * blockDim.x + threadIdx.x;
  if (l < L2NODE) lb[l] = batch[ue0[m0[l]]];
}

DEV unsigned encf(float x) {
  unsigned u = __float_as_uint(x);
  return (u & 0x80000000u) ? ~u : (u | 0x80000000u);
}
DEV float decf(unsigned e) {
  return (e & 0x80000000u) ? __uint_as_float(e & 0x7FFFFFFFu) : __uint_as_float(~e);
}

__global__ void pool_scatter(unsigned* __restrict__ mn, unsigned* __restrict__ mx,
                             float* __restrict__ sum, float* __restrict__ cnt,
                             const float* __restrict__ X, const int* __restrict__ idx,
                             int rows) {
  int i = blockIdx.x * blockDim.x + threadIdx.x;
  if (i >= rows * HID) return;
  int r = i >> 7, c = i & 127;
  int g = idx[r];
  float v = X[(size_t)r * HID + c];
  unsigned e = encf(v);
  int slot = g * HID + c;
  atomicMin(&mn[slot], e);
  atomicMax(&mx[slot], e);
  atomicAdd(&sum[slot], v);
  if (c == 0) atomicAdd(&cnt[g], 1.0f);
}
__global__ void pool_final(float* __restrict__ Hp, const unsigned* __restrict__ mn,
                           const unsigned* __restrict__ mx, const float* __restrict__ sum,
                           const float* __restrict__ cnt, int po) {
  int i = blockIdx.x * blockDim.x + threadIdx.x;
  if (i >= NGRAPH * HID) return;
  int g = i >> 7, c = i & 127;
  float fmn = decf(mn[i]); if (!(fabsf(fmn) < 3.0e38f)) fmn = 0.f;  // drop +-inf / nan
  float fmx = decf(mx[i]); if (!(fabsf(fmx) < 3.0e38f)) fmx = 0.f;
  float cn = cnt[g];
  float mean = sum[i] / (cn > 1.f ? cn : 1.f);
  float* row = Hp + (size_t)g * 1152 + po;
  row[c] = fmn; row[128 + c] = fmx; row[256 + c] = mean;
}

// =====================================================================
//  Final dot:  out[g] = h2[g,:] . fo_w + fo_b
// =====================================================================
__global__ void out_reduce(float* __restrict__ out, const float* __restrict__ h,
                           const float* __restrict__ w, const float* __restrict__ b) {
  int lane = threadIdx.x & 31;
  int g = blockIdx.x * (blockDim.x >> 5) + (threadIdx.x >> 5);
  if (g >= NGRAPH) return;
  float s = 0.f;
#pragma unroll
  for (int j = 0; j < 4; ++j) { int c = lane + 32 * j; s += h[(size_t)g * HID + c] * w[c]; }
#pragma unroll
  for (int o = 16; o > 0; o >>= 1) s += __shfl_xor(s, o);
  if (lane == 0) out[g] = s + b[0];
}

// =====================================================================
//  Host side
// =====================================================================
static inline int cdiv_i(long long a, long long b) { return (int)((a + b - 1) / b); }

static void launch_gemm(float* C, const float* A, const float* W, const float* bias,
                        int M, int N, int K, int mode, hipStream_t st) {
  const int tiles  = (M / 16) * (N / 32);
  const int blocks = cdiv_i(tiles, 8);
  if (mode == 1) {                       // only used with K=128 (ff1 + relu)
    gemm_wmma<128, 1><<<blocks, 256, 0, st>>>(C, A, W, bias, M, N);
    return;
  }
  switch (K) {
    case 16:   gemm_wmma<16,   0><<<blocks, 256, 0, st>>>(C, A, W, bias, M, N); break;
    case 128:  gemm_wmma<128,  0><<<blocks, 256, 0, st>>>(C, A, W, bias, M, N); break;
    case 256:  gemm_wmma<256,  0><<<blocks, 256, 0, st>>>(C, A, W, bias, M, N); break;
    case 1152: gemm_wmma<1152, 0><<<blocks, 256, 0, st>>>(C, A, W, bias, M, N); break;
    default: break;                      // unreachable for this pipeline
  }
}

extern "C" void kernel_launch(void* const* d_in, const int* in_sizes, int n_in,
                              void* d_out, int out_size, void* d_ws, size_t ws_size,
                              hipStream_t stream) {
  const float* x         = (const float*)d_in[0];
  const float* edge_attr = (const float*)d_in[1];
  const int*   ei        = (const int*)d_in[2];   // [2,E]
  const int*   ue        = (const int*)d_in[3];   // [2,EU]
  const int*   batch     = (const int*)d_in[4];
  const int*   l2map     = (const int*)d_in[5];   // [2,L2]
  const int*   l2ei      = (const int*)d_in[6];   // [2,EL]
  const float* lin_w     = (const float*)d_in[8];
  const float* lin_b     = (const float*)d_in[9];
  const float* g1w       = (const float*)d_in[10];
  const float* g1b       = (const float*)d_in[11];
  const float* g2w       = (const float*)d_in[12];
  const float* g2b       = (const float*)d_in[13];
  const float* t_in_w    = (const float*)d_in[14];
  const float* t_in_b    = (const float*)d_in[15];
  const float* t_out_w   = (const float*)d_in[16];
  const float* t_out_b   = (const float*)d_in[17];
  const float* ln1g      = (const float*)d_in[18];
  const float* ln1b      = (const float*)d_in[19];
  const float* ln2g      = (const float*)d_in[20];
  const float* ln2b      = (const float*)d_in[21];
  const float* ff1w      = (const float*)d_in[22];
  const float* ff1b      = (const float*)d_in[23];
  const float* ff2w      = (const float*)d_in[24];
  const float* ff2b      = (const float*)d_in[25];
  const float* f0w       = (const float*)d_in[26];
  const float* f0b       = (const float*)d_in[27];
  const float* f0g       = (const float*)d_in[28];
  const float* f0be      = (const float*)d_in[29];
  const float* f1w       = (const float*)d_in[30];
  const float* f1b       = (const float*)d_in[31];
  const float* f1g       = (const float*)d_in[32];
  const float* f1be      = (const float*)d_in[33];
  const float* fow       = (const float*)d_in[34];
  const float* fob       = (const float*)d_in[35];
  float* out = (float*)d_out;

  // ---- workspace arena (floats) ----
  float* ws = (float*)d_ws;
  size_t o = 0;
  auto alloc = [&](size_t n) { float* p = ws + o; o += n; return p; };
  float* H0e  = alloc((size_t)EUNIQ  * HID);   // edge linear output
  float* XWb  = alloc((size_t)L2NODE * HID);   // x@W for both GCNs
  float* H0   = alloc((size_t)NNODES * HID);
  float* xs   = alloc((size_t)NTOK   * HID);   // transformer activations
  float* bufQ = alloc((size_t)NTOK   * 384);   // qkv / ff hidden
  float* bufA = alloc((size_t)NTOK   * HID);   // attention output
  float* bufP = alloc((size_t)NTOK   * HID);   // proj / ff2 output
  float* H1   = alloc((size_t)L2NODE * HID);
  float* H2   = alloc((size_t)L2NODE * HID);
  float* deg  = alloc(L2NODE);                 // max(N, L2)
  int*   l2b  = (int*)alloc(L2NODE);
  unsigned* mnP[3]; unsigned* mxP[3]; float* smP[3]; float* cnP[3];
  for (int p = 0; p < 3; ++p) {
    mnP[p] = (unsigned*)alloc((size_t)NGRAPH * HID);
    mxP[p] = (unsigned*)alloc((size_t)NGRAPH * HID);
    smP[p] = alloc((size_t)NGRAPH * HID);
    cnP[p] = alloc(NGRAPH);
  }
  float* Hp = alloc((size_t)NGRAPH * 1152);
  float* t0 = alloc((size_t)NGRAPH * HID);
  float* t1 = alloc((size_t)NGRAPH * HID);

  const unsigned ENC_PINF = 0xFF800000u;  // encf(+inf)
  const unsigned ENC_NINF = 0x007FFFFFu;  // encf(-inf)
  const int TB = 256;

  // 1) H0_edge = edge_attr @ lin_w^T + lin_b            [EU,128], K=16
  launch_gemm(H0e, edge_attr, lin_w, lin_b, EUNIQ, HID, 16, 0, stream);

  // 2) GCN1: XW = x @ g1w^T; deg/dinv; aggregate
  launch_gemm(XWb, x, g1w, nullptr, NNODES, HID, HID, 0, stream);
  fill_f32<<<cdiv_i(NNODES, TB), TB, 0, stream>>>(deg, NNODES, 1.0f);  // self-loop
  deg_scatter<<<cdiv_i(EEDGES, TB), TB, 0, stream>>>(deg, ei + EEDGES, EEDGES);
  rsqrt_ip<<<cdiv_i(NNODES, TB), TB, 0, stream>>>(deg, NNODES);
  gcn_init<<<cdiv_i((long long)NNODES * HID, TB), TB, 0, stream>>>(H0, XWb, deg, g1b, NNODES);
  gcn_scatter<<<cdiv_i((long long)EEDGES * 32, TB), TB, 0, stream>>>(
      H0, XWb, deg, ei, ei + EEDGES, EEDGES);

  // 3) gather token sets -> xs [L2,6,128]
  gather_set<<<cdiv_i((long long)L2NODE * HID, TB), TB, 0, stream>>>(
      xs, H0, H0e, l2map, l2map + L2NODE, ue, ue + EUNIQ);

  // 4) transformer encoder (2 layers, post-norm, relu FF)
  for (int l = 0; l < NLAYER; ++l) {
    const float* inw = t_in_w  + (size_t)l * 384 * HID;
    const float* inb = t_in_b  + (size_t)l * 384;
    const float* ow  = t_out_w + (size_t)l * HID * HID;
    const float* ob  = t_out_b + (size_t)l * HID;
    const float* w1  = ff1w    + (size_t)l * 256 * HID;
    const float* b1  = ff1b    + (size_t)l * 256;
    const float* w2  = ff2w    + (size_t)l * HID * 256;
    const float* b2  = ff2b    + (size_t)l * HID;
    launch_gemm(bufQ, xs, inw, inb, NTOK, 384, HID, 0, stream);           // qkv
    attn_kernel<<<cdiv_i((long long)L2NODE * NHEAD, TB), TB, 0, stream>>>(bufA, bufQ);
    launch_gemm(bufP, bufA, ow, ob, NTOK, HID, HID, 0, stream);           // out proj
    row_ln<<<cdiv_i((long long)NTOK, 8), TB, 0, stream>>>(
        xs, bufP, ln1g + (size_t)l * HID, ln1b + (size_t)l * HID, NTOK, 0);
    launch_gemm(bufQ, xs, w1, b1, NTOK, 256, HID, 1, stream);             // ff1 + relu
    launch_gemm(bufP, bufQ, w2, b2, NTOK, HID, 256, 0, stream);           // ff2
    row_ln<<<cdiv_i((long long)NTOK, 8), TB, 0, stream>>>(
        xs, bufP, ln2g + (size_t)l * HID, ln2b + (size_t)l * HID, NTOK, 0);
  }

  // 5) H1 = mean over tokens
  mean6<<<cdiv_i((long long)L2NODE * HID, TB), TB, 0, stream>>>(H1, xs);

  // 6) GCN2 + relu
  launch_gemm(XWb, H1, g2w, nullptr, L2NODE, HID, HID, 0, stream);
  fill_f32<<<cdiv_i(L2NODE, TB), TB, 0, stream>>>(deg, L2NODE, 1.0f);
  deg_scatter<<<cdiv_i(ELEDGE, TB), TB, 0, stream>>>(deg, l2ei + ELEDGE, ELEDGE);
  rsqrt_ip<<<cdiv_i(L2NODE, TB), TB, 0, stream>>>(deg, L2NODE);
  gcn_init<<<cdiv_i((long long)L2NODE * HID, TB), TB, 0, stream>>>(H2, XWb, deg, g2b, L2NODE);
  gcn_scatter<<<cdiv_i((long long)ELEDGE * 32, TB), TB, 0, stream>>>(
      H2, XWb, deg, l2ei, l2ei + ELEDGE, ELEDGE);
  relu_ip<<<cdiv_i((long long)L2NODE * HID, TB), TB, 0, stream>>>(H2, (size_t)L2NODE * HID);

  // 7) l2_batch
  l2batch_kernel<<<cdiv_i(L2NODE, TB), TB, 0, stream>>>(l2b, batch, ue, l2map);

  // 8) pooling -> Hp [G, 1152]
  const float* Xp[3]  = { H0, H1, H2 };
  const int*   Ip[3]  = { batch, l2b, l2b };
  const int    Rp[3]  = { NNODES, L2NODE, L2NODE };
  for (int p = 0; p < 3; ++p) {
    fill_u32<<<cdiv_i((long long)NGRAPH * HID, TB), TB, 0, stream>>>(mnP[p], (size_t)NGRAPH * HID, ENC_PINF);
    fill_u32<<<cdiv_i((long long)NGRAPH * HID, TB), TB, 0, stream>>>(mxP[p], (size_t)NGRAPH * HID, ENC_NINF);
    fill_f32<<<cdiv_i((long long)NGRAPH * HID, TB), TB, 0, stream>>>(smP[p], (size_t)NGRAPH * HID, 0.f);
    fill_f32<<<cdiv_i(NGRAPH, TB), TB, 0, stream>>>(cnP[p], NGRAPH, 0.f);
    pool_scatter<<<cdiv_i((long long)Rp[p] * HID, TB), TB, 0, stream>>>(
        mnP[p], mxP[p], smP[p], cnP[p], Xp[p], Ip[p], Rp[p]);
    pool_final<<<cdiv_i((long long)NGRAPH * HID, TB), TB, 0, stream>>>(
        Hp, mnP[p], mxP[p], smP[p], cnP[p], p * 384);
  }

  // 9) MLP head
  launch_gemm(t0, Hp, f0w, f0b, NGRAPH, HID, 1152, 0, stream);
  row_ln<<<cdiv_i(NGRAPH, 8), TB, 0, stream>>>(t0, nullptr, f0g, f0be, NGRAPH, 1);
  launch_gemm(t1, t0, f1w, f1b, NGRAPH, HID, HID, 0, stream);
  row_ln<<<cdiv_i(NGRAPH, 8), TB, 0, stream>>>(t1, nullptr, f1g, f1be, NGRAPH, 1);
  out_reduce<<<cdiv_i(NGRAPH, 8), TB, 0, stream>>>(out, t1, fow, fob);

  (void)in_sizes; (void)n_in; (void)out_size; (void)ws_size;
}
